// MainAttention_70446053589373
// MI455X (gfx1250) — compile-verified, matrix operand order checked by reading, not run on hardware
//
#include <hip/hip_runtime.h>

// ---------------- problem constants (from reference) ----------------
#define B_    4
#define N_    2048
#define C_    768
#define H_    12
#define HD_   64
#define M_TOK (B_ * N_)      // 8192 tokens
#define DQKV  (3 * C_)       // 2304
#define SCALE_ 0.125f        // HD^-0.5

// ---------------- WMMA types / helpers ----------------
typedef __attribute__((ext_vector_type(16))) __bf16       v16bf;
typedef __attribute__((ext_vector_type(8)))  float        v8f;
typedef __attribute__((ext_vector_type(4)))  unsigned int u32x4;

union FragBits { u32x4 u[2]; v16bf v; };

// Load the 16 bf16 values this lane owns of a 16x32 A-fragment (or the
// mirrored 32x16 B-fragment read from row-major B^T): per the CDNA5 WMMA
// 16-bit layout, each lane holds two 8-element K chunks 16 elements apart.
// Caller passes p = &src[row * ld + k0 + (lane>>4)*8]; must be 16B aligned.
__device__ __forceinline__ v16bf load_frag(const __bf16* p) {
  FragBits f;
  f.u[0] = *(const u32x4*)(p);
  f.u[1] = *(const u32x4*)(p + 16);
  return f.v;
}

__device__ __forceinline__ v8f wmma_bf16(v16bf a, v16bf b, v8f c) {
  return __builtin_amdgcn_wmma_f32_16x16x32_bf16(
      /*neg_a=*/false, a, /*neg_b=*/false, b,
      /*c_mod=*/(short)0, c, /*reuse_a=*/false, /*reuse_b=*/false);
}

__device__ __forceinline__ v8f vzero8() {
  v8f z = {0.f, 0.f, 0.f, 0.f, 0.f, 0.f, 0.f, 0.f};
  return z;
}

// Wave index as a compiler-provable SGPR value (threadIdx.x>>5 is wave-
// uniform, but the compiler can't prove it; readfirstlane scalarizes it so
// tile bases live in SGPRs and epilogue selects become scalar branches).
__device__ __forceinline__ int wave_id() {
  return __builtin_amdgcn_readfirstlane((int)(threadIdx.x >> 5));
}

// CDNA5 async copy global -> LDS (16B per lane), tracked by ASYNCcnt.
// GV addressing: lds_off = per-lane LDS byte offset (VDST VGPR), gaddr = 64-bit
// global address. Low 32 bits of a generic LDS pointer are the LDS offset
// (ISA 10.2 aperture rule).
__device__ __forceinline__ void async_load_b128(unsigned lds_off, const void* gaddr) {
  asm volatile("global_load_async_to_lds_b128 %0, %1, off"
               :: "v"(lds_off), "v"(gaddr) : "memory");
}
__device__ __forceinline__ void wait_asynccnt0() {
  asm volatile("s_wait_asynccnt 0x0" ::: "memory");
}

// ---------------- stage 0: fp32 -> bf16 ----------------
__global__ __launch_bounds__(256) void cvt_f32_bf16(const void* in_, void* out_, int n) {
  const float* in = (const float*)in_;
  __bf16* out = (__bf16*)out_;
  int i = blockIdx.x * blockDim.x + threadIdx.x;
  int stride = gridDim.x * blockDim.x;
  for (; i < n; i += stride) out[i] = (__bf16)in[i];
}

// ---------------- stage 1: QKV GEMM ----------------
// QKV[t, d] = sum_c X[t, c] * Wqkv[d, c]   (NT GEMM, both K-contiguous)
// Block = 128x128 tile, 8 waves, each wave a 32x64 tile (2x4 WMMA grid).
// Epilogue scatters to Q (scaled), K as [B,H,N,HD] and V^T as [B,H,HD,N].
__global__ __launch_bounds__(256)
void qkv_gemm(const void* X_, const void* W_, void* Q_, void* K_, void* VT_) {
  const __bf16* X = (const __bf16*)X_;
  const __bf16* W = (const __bf16*)W_;
  __bf16* Q = (__bf16*)Q_;
  __bf16* K = (__bf16*)K_;
  __bf16* VT = (__bf16*)VT_;

  const int lane = threadIdx.x & 31;   // wave32 (probe-confirmed)
  const int wave = wave_id();          // SGPR
  const int half = lane >> 4;
  const int lrow = lane & 15;
  const int bm = blockIdx.y * 128 + (wave >> 1) * 32;  // token rows (SGPR)
  const int bn = blockIdx.x * 128 + (wave & 1) * 64;   // qkv cols  (SGPR)

  v8f acc[2][4];
#pragma unroll
  for (int i = 0; i < 2; ++i)
#pragma unroll
    for (int j = 0; j < 4; ++j) acc[i][j] = vzero8();

  for (int k0 = 0; k0 < C_; k0 += 32) {
    v16bf a[2], b[4];
#pragma unroll
    for (int i = 0; i < 2; ++i)
      a[i] = load_frag(X + (size_t)(bm + i * 16 + lrow) * C_ + k0 + half * 8);
#pragma unroll
    for (int j = 0; j < 4; ++j)
      b[j] = load_frag(W + (size_t)(bn + j * 16 + lrow) * C_ + k0 + half * 8);
#pragma unroll
    for (int i = 0; i < 2; ++i)
#pragma unroll
      for (int j = 0; j < 4; ++j) acc[i][j] = wmma_bf16(a[i], b[j], acc[i][j]);
  }

  // Epilogue. A 16-wide col tile never crosses a q/k/v or head boundary
  // (both are 16-aligned), so `which` and `h` are SGPR-uniform -> scalar
  // branch, no per-lane exec-mask churn.
#pragma unroll
  for (int j = 0; j < 4; ++j) {
    const int dbase = bn + j * 16;          // multiple of 16, SGPR
    const int which = dbase / C_;           // 0=q 1=k 2=v   (SGPR)
    const int dd = dbase - which * C_;
    const int h = dd >> 6;                  // head           (SGPR)
    const int hd = (dd & (HD_ - 1)) + lrow; // lane's hd index
#pragma unroll
    for (int i = 0; i < 2; ++i) {
#pragma unroll
      for (int r = 0; r < 8; ++r) {
        float v = acc[i][j][r];
        int t = bm + i * 16 + r + half * 8; // C/D layout: row = r + 8*half
        int bidx = t >> 11;                 // t / N_
        int n = t & (N_ - 1);
        size_t bh = (size_t)(bidx * H_ + h);
        if (which == 0)
          Q[(bh * N_ + n) * HD_ + hd] = (__bf16)(v * SCALE_);
        else if (which == 1)
          K[(bh * N_ + n) * HD_ + hd] = (__bf16)v;
        else
          VT[(bh * HD_ + hd) * N_ + n] = (__bf16)v;
      }
    }
  }
}

// ---------------- stage 2: flash attention ----------------
// One block = 64 query rows of one (b,h). 4 waves; wave owns a 16-row stripe.
// Per key-block of 64: K and V^T tiles are staged once into LDS with CDNA5
// async-to-LDS copies (shared by all 4 waves), then WMMA operands come from
// LDS. Streaming softmax in f32 with wave32 shfl reductions.
__global__ __launch_bounds__(128)
void flash_attn(const void* Q_, const void* K_, const void* VT_, void* O_) {
  const __bf16* Q = (const __bf16*)Q_;
  const __bf16* K = (const __bf16*)K_;
  const __bf16* VT = (const __bf16*)VT_;
  __bf16* O = (__bf16*)O_;

  // padded stride 72 halves -> conflict-free ds_load_b128 across 16 rows
  __shared__ __align__(32) __bf16 ldsK[64 * 72];      // 64 keys x 64 hd
  __shared__ __align__(32) __bf16 ldsV[64 * 72];      // 64 hd  x 64 keys
  __shared__ __align__(32) __bf16 ldsP[4][16 * 72];   // wave-private P stripes

  const int tid = threadIdx.x;
  const int lane = tid & 31;
  const int wave = wave_id();          // SGPR
  const int half = lane >> 4;
  const int lrow = lane & 15;
  const int bh = blockIdx.y;           // b*H + h
  const int h = bh % H_;
  const int b = bh / H_;
  const int q0 = blockIdx.x * 64 + wave * 16;   // SGPR

  const __bf16* Qb = Q + (size_t)bh * N_ * HD_;
  const __bf16* Kb = K + (size_t)bh * N_ * HD_;
  const __bf16* Vb = VT + (size_t)bh * HD_ * N_;
  __bf16* myP = ldsP[wave];

  // async-copy work split: 8 threads per row, 16-byte chunks, 4 sweeps/tile
  const int srow = tid >> 3;           // 0..15
  const int scol = (tid & 7) * 8;      // half-element offset within row
  const unsigned ldsK_base = (unsigned)(unsigned long long)(&ldsK[0]);
  const unsigned ldsV_base = (unsigned)(unsigned long long)(&ldsV[0]);

  // Q fragments (rows fixed for whole kernel; Q already pre-scaled).
  v16bf aq0 = load_frag(Qb + (size_t)(q0 + lrow) * HD_ + 0 + half * 8);
  v16bf aq1 = load_frag(Qb + (size_t)(q0 + lrow) * HD_ + 32 + half * 8);

  float mi[8], li[8];
  v8f accO[4];
#pragma unroll
  for (int r = 0; r < 8; ++r) { mi[r] = -1e30f; li[r] = 0.f; }
#pragma unroll
  for (int jo = 0; jo < 4; ++jo) accO[jo] = vzero8();

  for (int kb = 0; kb < N_; kb += 64) {
    // ---- stage K (64x64) and V^T (64x64) tiles into LDS, async ----
#pragma unroll
    for (int s = 0; s < 4; ++s) {
      int row = srow + s * 16;
      async_load_b128(ldsK_base + (unsigned)(row * 72 + scol) * 2,
                      Kb + (size_t)(kb + row) * HD_ + scol);
      async_load_b128(ldsV_base + (unsigned)(row * 72 + scol) * 2,
                      Vb + (size_t)row * N_ + kb + scol);
    }
    wait_asynccnt0();
    __syncthreads();

    // ---- S = Q * K^T for this 16x64 stripe (K operand from LDS) ----
    v8f s[4];
#pragma unroll
    for (int j = 0; j < 4; ++j) {
      const __bf16* krow = ldsK + (j * 16 + lrow) * 72;
      v16bf bk0 = load_frag(krow + 0 + half * 8);
      v16bf bk1 = load_frag(krow + 32 + half * 8);
      v8f z = vzero8();
      z = wmma_bf16(aq0, bk0, z);
      z = wmma_bf16(aq1, bk1, z);
      s[j] = z;
    }

    // ---- online softmax; row = r + 8*half lives in the lane's 16-half ----
#pragma unroll
    for (int r = 0; r < 8; ++r) {
      float mx = fmaxf(fmaxf(s[0][r], s[1][r]), fmaxf(s[2][r], s[3][r]));
      mx = fmaxf(mx, __shfl_xor(mx, 1, 32));
      mx = fmaxf(mx, __shfl_xor(mx, 2, 32));
      mx = fmaxf(mx, __shfl_xor(mx, 4, 32));
      mx = fmaxf(mx, __shfl_xor(mx, 8, 32));
      float mnew = fmaxf(mi[r], mx);
      float sc = __expf(mi[r] - mnew);
      mi[r] = mnew;
      float rs = 0.f;
#pragma unroll
      for (int j = 0; j < 4; ++j) {
        float p = __expf(s[j][r] - mnew);
        s[j][r] = p;
        rs += p;
      }
      rs += __shfl_xor(rs, 1, 32);
      rs += __shfl_xor(rs, 2, 32);
      rs += __shfl_xor(rs, 4, 32);
      rs += __shfl_xor(rs, 8, 32);
      li[r] = li[r] * sc + rs;
#pragma unroll
      for (int jo = 0; jo < 4; ++jo) accO[jo][r] *= sc;
      // stash P row into the wave-private LDS stripe
      int row = r + half * 8;
#pragma unroll
      for (int j = 0; j < 4; ++j)
        myP[row * 72 + j * 16 + lrow] = (__bf16)s[j][r];
    }

    // ---- reload P as A-fragments (same-wave LDS, DS ops are in-order) ----
    v16bf ap0 = load_frag(myP + lrow * 72 + 0 + half * 8);
    v16bf ap1 = load_frag(myP + lrow * 72 + 32 + half * 8);

    // ---- O += P * V  (B operand from LDS V^T: hd rows, key-contiguous) ----
#pragma unroll
    for (int jo = 0; jo < 4; ++jo) {
      const __bf16* vrow = ldsV + (jo * 16 + lrow) * 72;
      v16bf bv0 = load_frag(vrow + 0 + half * 8);
      v16bf bv1 = load_frag(vrow + 32 + half * 8);
      accO[jo] = wmma_bf16(ap0, bv0, accO[jo]);
      accO[jo] = wmma_bf16(ap1, bv1, accO[jo]);
    }

    __syncthreads();  // tiles consumed; safe to overwrite next iteration
  }

  // ---- epilogue: O[b, n, h*HD + c] = accO / l ----
#pragma unroll
  for (int r = 0; r < 8; ++r) {
    float inv = 1.0f / li[r];
    int n = q0 + r + half * 8;
#pragma unroll
    for (int jo = 0; jo < 4; ++jo) {
      int c = h * HD_ + jo * 16 + lrow;
      O[((size_t)(b * N_ + n)) * C_ + c] = (__bf16)(accO[jo][r] * inv);
    }
  }
}

// ---------------- stage 3: output projection ----------------
// out[t, d] = sum_c O[t, c] * Wproj[d, c] + bias[d]   (fp32 out)
__global__ __launch_bounds__(256)
void proj_gemm(const void* A_, const void* W_, const float* __restrict__ bias,
               float* __restrict__ out) {
  const __bf16* A = (const __bf16*)A_;
  const __bf16* W = (const __bf16*)W_;

  const int lane = threadIdx.x & 31;
  const int wave = wave_id();          // SGPR
  const int half = lane >> 4;
  const int lrow = lane & 15;
  const int bm = blockIdx.y * 128 + (wave >> 1) * 32;
  const int bn = blockIdx.x * 128 + (wave & 1) * 64;

  v8f acc[2][4];
#pragma unroll
  for (int i = 0; i < 2; ++i)
#pragma unroll
    for (int j = 0; j < 4; ++j) acc[i][j] = vzero8();

  for (int k0 = 0; k0 < C_; k0 += 32) {
    v16bf a[2], b[4];
#pragma unroll
    for (int i = 0; i < 2; ++i)
      a[i] = load_frag(A + (size_t)(bm + i * 16 + lrow) * C_ + k0 + half * 8);
#pragma unroll
    for (int j = 0; j < 4; ++j)
      b[j] = load_frag(W + (size_t)(bn + j * 16 + lrow) * C_ + k0 + half * 8);
#pragma unroll
    for (int i = 0; i < 2; ++i)
#pragma unroll
      for (int j = 0; j < 4; ++j) acc[i][j] = wmma_bf16(a[i], b[j], acc[i][j]);
  }

#pragma unroll
  for (int i = 0; i < 2; ++i) {
#pragma unroll
    for (int j = 0; j < 4; ++j) {
#pragma unroll
      for (int r = 0; r < 8; ++r) {
        int t = bm + i * 16 + r + half * 8;
        int d = bn + j * 16 + lrow;
        out[(size_t)t * C_ + d] = acc[i][j][r] + bias[d];
      }
    }
  }
}

// ---------------- launcher ----------------
extern "C" void kernel_launch(void* const* d_in, const int* in_sizes, int n_in,
                              void* d_out, int out_size, void* d_ws, size_t ws_size,
                              hipStream_t stream) {
  const float* x      = (const float*)d_in[0];  // [B, N, C]
  const float* w_qkv  = (const float*)d_in[1];  // [3C, C]
  const float* w_proj = (const float*)d_in[2];  // [C, C]
  const float* b_proj = (const float*)d_in[3];  // [C]
  float* out = (float*)d_out;                   // [B, N, C] fp32

  char* ws = (char*)d_ws;
  size_t off = 0;
  auto take = [&](size_t elems) -> void* {
    void* p = (void*)(ws + off);
    off += (elems * 2 + 255) & ~(size_t)255;   // bf16 elements, 256B aligned
    return p;
  };
  void* xb  = take((size_t)M_TOK * C_);         // x in bf16
  void* wqb = take((size_t)DQKV * C_);          // w_qkv bf16
  void* wpb = take((size_t)C_ * C_);            // w_proj bf16
  void* qb  = take((size_t)B_ * H_ * N_ * HD_); // Q  [B,H,N,HD] (pre-scaled)
  void* kb  = take((size_t)B_ * H_ * N_ * HD_); // K  [B,H,N,HD]
  void* vtb = take((size_t)B_ * H_ * HD_ * N_); // V^T [B,H,HD,N]
  void* ob  = take((size_t)M_TOK * C_);         // attention output [B,N,C]

  cvt_f32_bf16<<<2048, 256, 0, stream>>>(x, xb, M_TOK * C_);
  cvt_f32_bf16<<<2048, 256, 0, stream>>>(w_qkv, wqb, DQKV * C_);
  cvt_f32_bf16<<<512, 256, 0, stream>>>(w_proj, wpb, C_ * C_);

  qkv_gemm<<<dim3(DQKV / 128, M_TOK / 128), 256, 0, stream>>>(xb, wqb, qb, kb, vtb);
  flash_attn<<<dim3(N_ / 64, B_ * H_), 128, 0, stream>>>(qb, kb, vtb, ob);
  proj_gemm<<<dim3(C_ / 128, M_TOK / 128), 256, 0, stream>>>(ob, wpb, b_proj, out);
}
